// DecoderRNNWithAttention_82875688944305
// MI455X (gfx1250) — matrix operand order repaired and checked by reading, not compiled, not run on hardware
//
#include <hip/hip_runtime.h>
#include <hip/hip_bf16.h>

// ---------------------------------------------------------------------------
// MI455X (gfx1250) implementation of the attention-LSTM caption decoder.
// All GEMMs via v_wmma_f32_16x16x32_bf16 (wave32). Dominant cost is the
// per-step vocab projection (128 x 30000 x 512): weights kept bf16 + L2
// resident, outputs staged through LDS for coalesced f32 stores (HBM bound).
// ---------------------------------------------------------------------------

#define BATCH 128
#define NPIX  196
#define EENC_ 512
#define EDIM  512
#define HDIM  512
#define ADIM  512
#define VOCAB 30000
#define VPAD  30080        // 235 * 128
#define LCAP  20
#define TSTEP 19
#define KCAT  1536         // E + EENC + H
#define NCAT  2048         // 4H

typedef __bf16 bf16_t;
typedef __attribute__((ext_vector_type(16))) __bf16 v16bf;
typedef __attribute__((ext_vector_type(8)))  __bf16 v8bf;
typedef __attribute__((ext_vector_type(8)))  float  v8f;

union Frag { v16bf v; v8bf h[2]; };

static __device__ __forceinline__ float bf2f(bf16_t x) { return (float)x; }
static __device__ __forceinline__ bf16_t f2bf(float x) { return (bf16_t)x; }
static __device__ __forceinline__ float sigm(float x) { return 1.0f / (1.0f + __expf(-x)); }

// ---------------------------------------------------------------- sort (B=128)
__global__ __launch_bounds__(128) void k_sort(
    const int* __restrict__ lengths, const int* __restrict__ captions,
    int* __restrict__ si, int* __restrict__ dl,
    float* __restrict__ out_cap, float* __restrict__ out_si, float* __restrict__ out_dl)
{
    __shared__ int s_len[BATCH];
    int i = threadIdx.x;
    s_len[i] = lengths[i];
    __syncthreads();
    int li = s_len[i];
    int rank = 0;
    for (int j = 0; j < BATCH; ++j) {
        int lj = s_len[j];
        if (lj > li || (lj == li && j < i)) rank++;   // stable descending
    }
    si[rank] = i;
    dl[rank] = li - 1;
    out_si[rank] = (float)i;
    out_dl[rank] = (float)(li - 1);
    __syncthreads();
    int src = si[i];
    for (int l = 0; l < LCAP; ++l)
        out_cap[i * LCAP + l] = (float)captions[src * LCAP + l];
}

// ---------------------------------------------------------------- utilities
__global__ void k_zero(float* __restrict__ p, long n) {
    long i = (long)blockIdx.x * blockDim.x + threadIdx.x;
    if (i < n) p[i] = 0.0f;
}

__global__ void k_f32_to_bf16(const float* __restrict__ s, bf16_t* __restrict__ d, long n) {
    long i = (long)blockIdx.x * blockDim.x + threadIdx.x;
    if (i < n) d[i] = f2bf(s[i]);
}

__global__ void k_convert_pad_vocab(const float* __restrict__ W, bf16_t* __restrict__ d) {
    long i = (long)blockIdx.x * blockDim.x + threadIdx.x;
    if (i >= (long)VPAD * HDIM) return;
    long r = i / HDIM;
    d[i] = (r < VOCAB) ? f2bf(W[i]) : f2bf(0.0f);
}

__global__ void k_build_wcat(const float* __restrict__ Wih, const float* __restrict__ Whh,
                             bf16_t* __restrict__ Wcat) {
    long i = (long)blockIdx.x * blockDim.x + threadIdx.x;
    if (i >= (long)NCAT * KCAT) return;
    long n = i / KCAT, k = i % KCAT;
    float v = (k < 1024) ? Wih[n * 1024 + k] : Whh[n * 512 + (k - 1024)];
    Wcat[i] = f2bf(v);
}

__global__ void k_bcat(const float* __restrict__ bih, const float* __restrict__ bhh,
                       float* __restrict__ bcat) {
    int i = blockIdx.x * blockDim.x + threadIdx.x;
    if (i < NCAT) bcat[i] = bih[i] + bhh[i];
}

__global__ void k_emb(const float* __restrict__ embW, const int* __restrict__ captions,
                      bf16_t* __restrict__ emb) {
    long i = (long)blockIdx.x * blockDim.x + threadIdx.x;
    if (i >= (long)BATCH * LCAP * EDIM) return;
    long b = i / (LCAP * EDIM);
    long le = i % (LCAP * EDIM);
    long l = le / EDIM, e = le % EDIM;
    int cap = captions[b * LCAP + l];
    emb[i] = f2bf(embW[(long)cap * EDIM + e]);
}

__global__ void k_feats(const float* __restrict__ features, const int* __restrict__ si,
                        bf16_t* __restrict__ fb) {
    long i = (long)blockIdx.x * blockDim.x + threadIdx.x;
    if (i >= (long)BATCH * NPIX * EENC_) return;
    long b = i / (NPIX * EENC_);
    long pe = i % (NPIX * EENC_);
    fb[i] = f2bf(features[(long)si[b] * NPIX * EENC_ + pe]);
}

__global__ void k_mean(const float* __restrict__ features, const int* __restrict__ si,
                       float* __restrict__ mf) {
    int i = blockIdx.x * blockDim.x + threadIdx.x;
    if (i >= BATCH * EENC_) return;
    int b = i >> 9, e = i & 511;
    const float* base = features + (long)si[b] * NPIX * EENC_ + e;
    float s = 0.0f;
    for (int p = 0; p < NPIX; ++p) s += base[(long)p * EENC_];
    mf[i] = s * (1.0f / (float)NPIX);
}

__global__ void k_init_hc(const float* __restrict__ mf,
                          const float* __restrict__ Wh, const float* __restrict__ bh,
                          const float* __restrict__ Wc, const float* __restrict__ bc,
                          float* __restrict__ h, float* __restrict__ c, bf16_t* __restrict__ hbf) {
    int i = blockIdx.x * blockDim.x + threadIdx.x;
    if (i >= BATCH * HDIM) return;
    int b = i >> 9, j = i & 511;
    const float* m = mf + b * EENC_;
    const float* wh = Wh + (long)j * EENC_;
    const float* wc = Wc + (long)j * EENC_;
    float sh = bh[j], sc = bc[j];
    for (int k = 0; k < EENC_; ++k) { float mv = m[k]; sh += mv * wh[k]; sc += mv * wc[k]; }
    h[i] = sh; c[i] = sc; hbf[i] = f2bf(sh);
}

// ------------------------------------------------------- generic WMMA GEMM
// C[M x Ntrue] = A[M x K](bf16, lda) * W[N x K]^T(bf16, ldw) + bias
// mode 0: f32 out (Cf, ldc)   mode 1: bf16 out (Cb, ldc)
// mode 2: masked write to predictions dout[((b*T)+t)*V + v]
// Block = 256 threads = 8 waves; wave w owns one 16x16 N-tile; grid.x covers
// N in chunks of 128, grid.y covers M in chunks of 16. W rows padded to grid.
__global__ __launch_bounds__(256) void k_gemm_wmma(
    const bf16_t* __restrict__ Amat, int lda,
    const bf16_t* __restrict__ Wmat, int ldw,
    const float* __restrict__ bias,
    float* __restrict__ Cf, bf16_t* __restrict__ Cb, float* __restrict__ dout,
    int ldc, int Ntrue, int Kdim, int mode,
    const int* __restrict__ declen, int t)
{
    __shared__ float lds_c[8][16][17];
    const int tid  = threadIdx.x;
    const int lane = tid & 31;
    const int w    = tid >> 5;
    const int grp  = lane >> 4;
    const int mn   = lane & 15;
    const int m0   = blockIdx.y * 16;
    const int n0   = (blockIdx.x * 8 + w) * 16;

    // A fragment base: lanes 0-15 -> K {0..7,16..23}, lanes 16-31 -> {8..15,24..31}
    const bf16_t* aBase = Amat + (size_t)(m0 + mn) * lda + grp * 8;
    // B fragment base: lanes 0-15 -> K {0..7,16..23}, lanes 16-31 -> {8..15,24..31}
    const bf16_t* wBase = Wmat + (size_t)(n0 + mn) * ldw + grp * 8;

    v8f acc = {};
    for (int k0 = 0; k0 < Kdim; k0 += 32) {
        Frag a, b;
        a.h[0] = *(const v8bf*)(aBase + k0);
        a.h[1] = *(const v8bf*)(aBase + k0 + 16);
        b.h[0] = *(const v8bf*)(wBase + k0);
        b.h[1] = *(const v8bf*)(wBase + k0 + 16);
        __builtin_prefetch((const void*)(wBase + k0 + 128), 0, 1);
        acc = __builtin_amdgcn_wmma_f32_16x16x32_bf16(
            false, a.v, false, b.v, (short)0, acc, false, false);
    }

    // spill 16x16 f32 tile: VGPR r -> row (r + 8*grp), col mn
#pragma unroll
    for (int r = 0; r < 8; ++r)
        lds_c[w][r + 8 * grp][mn] = acc[r];
    __syncthreads();

    // coalesced store of the block's 16 x 128 region
    const int nblk = blockIdx.x * 128;
    const int row  = tid >> 4;
    const int cb   = (tid & 15) * 8;
#pragma unroll
    for (int j = 0; j < 8; ++j) {
        int col = cb + j;
        int v   = nblk + col;
        if (v >= Ntrue) continue;
        float val = lds_c[col >> 4][row][col & 15];
        int b = m0 + row;
        if (mode == 0) {
            Cf[(size_t)b * ldc + v] = val + bias[v];
        } else if (mode == 1) {
            Cb[(size_t)b * ldc + v] = f2bf(val + bias[v]);
        } else {
            bool msk = declen[b] > t;
            dout[((size_t)b * TSTEP + t) * VOCAB + v] = msk ? (val + bias[v]) : 0.0f;
        }
    }
}

// ------------------------------------------------- fused attention (1 blk/row)
__global__ __launch_bounds__(256) void k_attention(
    const bf16_t* __restrict__ att1, const float* __restrict__ att2,
    const bf16_t* __restrict__ fb, const float* __restrict__ gate_pre,
    const float* __restrict__ Wfull, const float* __restrict__ bfull,
    const bf16_t* __restrict__ emb, const bf16_t* __restrict__ hbf,
    const int* __restrict__ declen, int t,
    bf16_t* __restrict__ xh, float* __restrict__ alphas_out)
{
    __shared__ float s_att2[ADIM];
    __shared__ float s_wf[ADIM];
    __shared__ float s_red[256];
    __shared__ float s_alpha[NPIX];
    const int b = blockIdx.x, tid = threadIdx.x;

    for (int i = tid; i < ADIM; i += 256) {
        s_att2[i] = att2[b * ADIM + i];
        s_wf[i]   = Wfull[i];
    }
    __syncthreads();

    float sc = -1e30f;
    if (tid < NPIX) {
        const bf16_t* row = att1 + ((size_t)b * NPIX + tid) * ADIM;
        float s = bfull[0];
        for (int a = 0; a < ADIM; ++a) {
            float v = bf2f(row[a]) + s_att2[a];
            s += s_wf[a] * fmaxf(v, 0.0f);
        }
        sc = s;
    }
    s_red[tid] = sc;
    __syncthreads();
    for (int off = 128; off > 0; off >>= 1) {
        if (tid < off) s_red[tid] = fmaxf(s_red[tid], s_red[tid + off]);
        __syncthreads();
    }
    float mx = s_red[0];
    __syncthreads();
    float ex = (tid < NPIX) ? __expf(sc - mx) : 0.0f;
    s_red[tid] = ex;
    __syncthreads();
    for (int off = 128; off > 0; off >>= 1) {
        if (tid < off) s_red[tid] += s_red[tid + off];
        __syncthreads();
    }
    float inv = 1.0f / s_red[0];
    if (tid < NPIX) s_alpha[tid] = ex * inv;
    __syncthreads();

    bool msk = declen[b] > t;
    if (tid < NPIX)
        alphas_out[((size_t)b * LCAP + t) * NPIX + tid] = msk ? s_alpha[tid] : 0.0f;

    // awe = alpha . feats, gate, and build xh = [emb_t | gate*awe | h]
    for (int e = tid; e < EENC_; e += 256) {
        float awe = 0.0f;
        const bf16_t* fc = fb + (size_t)b * NPIX * EENC_ + e;
        for (int p = 0; p < NPIX; ++p) awe += s_alpha[p] * bf2f(fc[(size_t)p * EENC_]);
        awe *= sigm(gate_pre[b * EENC_ + e]);
        xh[(size_t)b * KCAT + e]              = emb[((size_t)b * LCAP + t) * EDIM + e];
        xh[(size_t)b * KCAT + EDIM + e]       = f2bf(awe);
        xh[(size_t)b * KCAT + EDIM + EENC_ + e] = hbf[b * HDIM + e];
    }
}

// ----------------------------------------------------------- LSTM pointwise
__global__ void k_lstm(const float* __restrict__ g, float* __restrict__ h,
                       float* __restrict__ c, bf16_t* __restrict__ hbf,
                       bf16_t* __restrict__ hnewbf, const int* __restrict__ declen, int t)
{
    int i = blockIdx.x * blockDim.x + threadIdx.x;
    if (i >= BATCH * HDIM) return;
    int b = i >> 9, j = i & 511;
    const float* gr = g + (size_t)b * NCAT;
    float gi = gr[j], gf = gr[512 + j], gg = gr[1024 + j], go = gr[1536 + j];
    float cn = sigm(gf) * c[i] + sigm(gi) * tanhf(gg);
    float hn = sigm(go) * tanhf(cn);
    hnewbf[i] = f2bf(hn);
    bool m = declen[b] > t;
    float hu = m ? hn : h[i];
    float cu = m ? cn : c[i];
    h[i] = hu; c[i] = cu; hbf[i] = f2bf(hu);
}

// ---------------------------------------------------------------------------
extern "C" void kernel_launch(void* const* d_in, const int* in_sizes, int n_in,
                              void* d_out, int out_size, void* d_ws, size_t ws_size,
                              hipStream_t stream) {
    (void)in_sizes; (void)n_in; (void)out_size; (void)ws_size;
    const float* features  = (const float*)d_in[0];
    const int*   captions  = (const int*)d_in[1];
    const int*   lengths   = (const int*)d_in[2];
    const float* embW      = (const float*)d_in[3];
    const float* W_enc_att = (const float*)d_in[4];
    const float* b_enc_att = (const float*)d_in[5];
    const float* W_dec_att = (const float*)d_in[6];
    const float* b_dec_att = (const float*)d_in[7];
    const float* W_full    = (const float*)d_in[8];
    const float* b_full    = (const float*)d_in[9];
    const float* W_lin     = (const float*)d_in[10];
    const float* b_lin     = (const float*)d_in[11];
    const float* W_init_h  = (const float*)d_in[12];
    const float* b_init_h  = (const float*)d_in[13];
    const float* W_init_c  = (const float*)d_in[14];
    const float* b_init_c  = (const float*)d_in[15];
    const float* W_fbeta   = (const float*)d_in[16];
    const float* b_fbeta   = (const float*)d_in[17];
    const float* W_ih      = (const float*)d_in[18];
    const float* W_hh      = (const float*)d_in[19];
    const float* b_ih      = (const float*)d_in[20];
    const float* b_hh      = (const float*)d_in[21];

    float* out = (float*)d_out;
    // output layout: predictions | captions_sorted | alphas | sort_ind | decode_len
    float* o_pred   = out;
    float* o_cap    = o_pred + (size_t)BATCH * TSTEP * VOCAB;
    float* o_alphas = o_cap + (size_t)BATCH * LCAP;
    float* o_si     = o_alphas + (size_t)BATCH * LCAP * NPIX;
    float* o_dl     = o_si + BATCH;

    // workspace carve-up (256B aligned)
    char* ws = (char*)d_ws;
    size_t off = 0;
    auto alloc = [&](size_t bytes) -> char* {
        char* p = ws + off;
        off = (off + bytes + 255) & ~(size_t)255;
        return p;
    };
    int*    w_si    = (int*)alloc(BATCH * sizeof(int));
    int*    w_dl    = (int*)alloc(BATCH * sizeof(int));
    float*  w_mean  = (float*)alloc((size_t)BATCH * EENC_ * 4);
    float*  w_h     = (float*)alloc((size_t)BATCH * HDIM * 4);
    float*  w_c     = (float*)alloc((size_t)BATCH * HDIM * 4);
    bf16_t* w_hbf   = (bf16_t*)alloc((size_t)BATCH * HDIM * 2);
    bf16_t* w_hnbf  = (bf16_t*)alloc((size_t)BATCH * HDIM * 2);
    float*  w_att2  = (float*)alloc((size_t)BATCH * ADIM * 4);
    float*  w_gate  = (float*)alloc((size_t)BATCH * EENC_ * 4);
    bf16_t* w_xh    = (bf16_t*)alloc((size_t)BATCH * KCAT * 2);
    float*  w_g     = (float*)alloc((size_t)BATCH * NCAT * 4);
    float*  w_bcat  = (float*)alloc((size_t)NCAT * 4);
    bf16_t* w_emb   = (bf16_t*)alloc((size_t)BATCH * LCAP * EDIM * 2);
    bf16_t* w_Wdec  = (bf16_t*)alloc((size_t)ADIM * HDIM * 2);
    bf16_t* w_Wfb   = (bf16_t*)alloc((size_t)EENC_ * HDIM * 2);
    bf16_t* w_Wenc  = (bf16_t*)alloc((size_t)ADIM * EENC_ * 2);
    bf16_t* w_Wcat  = (bf16_t*)alloc((size_t)NCAT * KCAT * 2);
    bf16_t* w_Wlin  = (bf16_t*)alloc((size_t)VPAD * HDIM * 2);
    bf16_t* w_fb    = (bf16_t*)alloc((size_t)BATCH * NPIX * EENC_ * 2);
    bf16_t* w_att1  = (bf16_t*)alloc((size_t)BATCH * NPIX * ADIM * 2);

    auto cdiv = [](long a, long b) { return (int)((a + b - 1) / b); };

    // ---- setup
    k_sort<<<1, 128, 0, stream>>>(lengths, captions, w_si, w_dl, o_cap, o_si, o_dl);
    {
        long n = (long)BATCH * LCAP * NPIX;
        k_zero<<<cdiv(n, 256), 256, 0, stream>>>(o_alphas, n);
    }
    k_f32_to_bf16<<<cdiv((long)ADIM * HDIM, 256), 256, 0, stream>>>(W_dec_att, w_Wdec, (long)ADIM * HDIM);
    k_f32_to_bf16<<<cdiv((long)EENC_ * HDIM, 256), 256, 0, stream>>>(W_fbeta, w_Wfb, (long)EENC_ * HDIM);
    k_f32_to_bf16<<<cdiv((long)ADIM * EENC_, 256), 256, 0, stream>>>(W_enc_att, w_Wenc, (long)ADIM * EENC_);
    k_convert_pad_vocab<<<cdiv((long)VPAD * HDIM, 256), 256, 0, stream>>>(W_lin, w_Wlin);
    k_build_wcat<<<cdiv((long)NCAT * KCAT, 256), 256, 0, stream>>>(W_ih, W_hh, w_Wcat);
    k_bcat<<<cdiv(NCAT, 256), 256, 0, stream>>>(b_ih, b_hh, w_bcat);
    k_emb<<<cdiv((long)BATCH * LCAP * EDIM, 256), 256, 0, stream>>>(embW, captions, w_emb);
    k_feats<<<cdiv((long)BATCH * NPIX * EENC_, 256), 256, 0, stream>>>(features, w_si, w_fb);
    k_mean<<<cdiv(BATCH * EENC_, 256), 256, 0, stream>>>(features, w_si, w_mean);
    k_init_hc<<<cdiv(BATCH * HDIM, 256), 256, 0, stream>>>(
        w_mean, W_init_h, b_init_h, W_init_c, b_init_c, w_h, w_c, w_hbf);

    // ---- one-time: att1 = feats @ W_enc_att^T + b   (25088 x 512, K=512) -> bf16
    k_gemm_wmma<<<dim3(ADIM / 128, BATCH * NPIX / 16), 256, 0, stream>>>(
        w_fb, EENC_, w_Wenc, EENC_, b_enc_att,
        nullptr, w_att1, nullptr, ADIM, ADIM, EENC_, 1, nullptr, 0);

    // ---- decode steps
    for (int t = 0; t < TSTEP; ++t) {
        // att2 = h @ W_dec_att^T + b
        k_gemm_wmma<<<dim3(ADIM / 128, BATCH / 16), 256, 0, stream>>>(
            w_hbf, HDIM, w_Wdec, HDIM, b_dec_att,
            w_att2, nullptr, nullptr, ADIM, ADIM, HDIM, 0, nullptr, 0);
        // gate_pre = h @ W_fbeta^T + b
        k_gemm_wmma<<<dim3(EENC_ / 128, BATCH / 16), 256, 0, stream>>>(
            w_hbf, HDIM, w_Wfb, HDIM, b_fbeta,
            w_gate, nullptr, nullptr, EENC_, EENC_, HDIM, 0, nullptr, 0);
        // softmax attention, awe, gating, xh = [emb_t | awe | h]
        k_attention<<<BATCH, 256, 0, stream>>>(
            w_att1, w_att2, w_fb, w_gate, W_full, b_full, w_emb, w_hbf,
            w_dl, t, w_xh, o_alphas);
        // g = xh @ [W_ih|W_hh]^T + (b_ih + b_hh)   (128 x 2048, K=1536)
        k_gemm_wmma<<<dim3(NCAT / 128, BATCH / 16), 256, 0, stream>>>(
            w_xh, KCAT, w_Wcat, KCAT, w_bcat,
            w_g, nullptr, nullptr, NCAT, NCAT, KCAT, 0, nullptr, 0);
        // LSTM pointwise
        k_lstm<<<cdiv(BATCH * HDIM, 256), 256, 0, stream>>>(
            w_g, w_h, w_c, w_hbf, w_hnbf, w_dl, t);
        // preds = h_new @ W_lin^T + b_lin, masked write (128 x 30000, K=512)
        k_gemm_wmma<<<dim3(VPAD / 128, BATCH / 16), 256, 0, stream>>>(
            w_hnbf, HDIM, w_Wlin, HDIM, b_lin,
            nullptr, nullptr, o_pred, VOCAB, VOCAB, HDIM, 2, w_dl, t);
    }
}
